// MambaBlock_15857019257117
// MI455X (gfx1250) — compile-verified, matrix operand order checked by reading, not run on hardware
//
#include <hip/hip_runtime.h>
#include <hip/hip_bf16.h>
#include <cstdint>

// ---------------------------------------------------------------------------
// Mamba block forward for MI455X (gfx1250, wave32).
//  * GEMMs: v_wmma_f32_16x16x32_bf16.  A tiles staged with per-lane
//    global_load_async_to_lds_b128 (ASYNCcnt); B tiles staged with the
//    Tensor Data Mover (tensor_load_to_lds, TENSORcnt) using a 2D D#
//    descriptor -- both CDNA5 async copy paths, double-buffered in LDS.
//  * Weights pre-transposed to bf16 [N][K] so tiles are contiguous copies.
//  * Selective scan: 16 states/thread in VGPRs, v_exp_f32 recurrence,
//    B_t/C_t broadcast via double-buffered LDS, wave32 shuffle reduction.
// ---------------------------------------------------------------------------

typedef __attribute__((ext_vector_type(16))) __bf16 bf16x16;
typedef __attribute__((ext_vector_type(8)))  __bf16 bf16x8;
typedef __attribute__((ext_vector_type(8)))  float  f32x8;
typedef unsigned int mb_u32x4 __attribute__((ext_vector_type(4)));
typedef int          mb_i32x4 __attribute__((ext_vector_type(4)));
typedef int          mb_i32x8 __attribute__((ext_vector_type(8)));

#define MB_BSZ     2
#define MB_L       2048
#define MB_DMODEL  1024
#define MB_DINNER  2048
#define MB_DSTATE  128
#define MB_DTRANK  64
#define MB_ROWS    (MB_BSZ * MB_L)            // 4096
#define MB_XPROJ_N (MB_DTRANK + 2*MB_DSTATE)  // 320
#define MB_XPROJ_NPAD 384                     // pad ragged N to tile multiple

// gfx1250 async copy: 16 bytes global -> LDS per lane, tracked by ASYNCcnt.
#define MB_ASYNC_CP16(ldsOff, gaddr)                                        \
  asm volatile("global_load_async_to_lds_b128 %0, %1, off"                  \
               :: "v"(ldsOff), "v"(gaddr) : "memory")
#define MB_WAIT_ASYNC() asm volatile("s_wait_asynccnt 0x0" ::: "memory")

__device__ __forceinline__ uint32_t mb_lds_off(const void* p) {
  // generic pointers to LDS carry the AS(3) offset in the low 32 bits
  return (uint32_t)(uintptr_t)p;
}

// WMMA 16-bit operand gather from an LDS [row][k] tile (row stride 32 bf16).
// lane<16 -> kbase 0, lane>=16 -> kbase 8; halves at [kbase,kbase+8) and
// [16+kbase,16+kbase+8): two 16-byte ds_load_b128 per fragment.
__device__ __forceinline__ bf16x16 mb_lds_frag(const __bf16* rowPtr, int kbase) {
  const bf16x8* p0 = reinterpret_cast<const bf16x8*>(rowPtr + kbase);
  const bf16x8* p1 = reinterpret_cast<const bf16x8*>(rowPtr + kbase + 16);
  bf16x8 lo = *p0;
  bf16x8 hi = *p1;
  bf16x16 f;
#pragma unroll
  for (int i = 0; i < 8; ++i) { f[i] = lo[i]; f[i + 8] = hi[i]; }
  return f;
}

__device__ __forceinline__ float mb_softplus(float v) {
  return (v > 20.0f) ? v : log1pf(__expf(v));
}

// ---------------------------------------------------------------------------
// bf16 WMMA GEMM:  C[M,N] = act(A[M,K] @ Bt[N,K]^T + bias)
//   A  : bf16 [M][K]  row-major, lda      (M%128==0, K%32==0 by construction)
//   Bt : bf16 [N*][K] row-major, ldb     (N* padded to 128; OOB rows valid mem)
//   C  : f32 [M][N] ldc;  Cb (optional): bf16 copy, ldcb
// Block 256 thr = 8 waves (4x2), tile 128x128, K-step 32, double-buffered LDS.
// A tile: per-lane 16B async copies.  B tile: one TDM descriptor per K-step
// (2D tile: 128 rows x 64B, row stride ldb elements), issued by wave 0.
// ---------------------------------------------------------------------------
__global__ __launch_bounds__(256)
void mb_gemm_bf16(const __bf16* __restrict__ A, const __bf16* __restrict__ Bt,
                  float* __restrict__ C, __bf16* __restrict__ Cb,
                  int M, int N, int K, int lda, int ldb, int ldc, int ldcb,
                  const float* __restrict__ bias, int act) {
  __shared__ __align__(16) __bf16 At[2][128][32];
  __shared__ __align__(16) __bf16 Bts[2][128][32];

  const int tid      = threadIdx.x;
  const int lane     = tid & 31;
  const int wave     = tid >> 5;
  const int wr       = wave >> 1;            // 0..3
  const int wc       = wave & 1;             // 0..1
  const int l16      = lane & 15;
  const int laneHalf = lane >> 4;
  const int kbase    = laneHalf * 8;

  const int blockM = blockIdx.y * 128;
  const int blockN = blockIdx.x * 128;

  // ---- A staging: 512 x 16B chunks per tile, 2 per thread ----
  const int srow = tid >> 2;                 // 0..63 (and +64)
  const int sq   = (tid & 3) * 8;            // k sub-offset (elements)
  const uint64_t ga0 = (uint64_t)(uintptr_t)(A + (size_t)(blockM + srow) * lda + sq);
  const uint64_t ga1 = ga0 + (uint64_t)64 * lda * sizeof(__bf16);
  const uint32_t la0 = mb_lds_off(&At[0][srow][sq]);

  // ---- B staging: TDM descriptor (ISA 8.3/8.4), groups 2/3 zero (2D) ----
  const uint32_t ldsB  = (uint32_t)__builtin_amdgcn_readfirstlane(
                             (int)mb_lds_off(&Bts[0][0][0]));
  const uint64_t gbB   = (uint64_t)(uintptr_t)(Bt + (size_t)blockN * ldb);
  mb_i32x8 g1d;
  g1d[0] = 0x10000;                               // data_size = 2 bytes
  g1d[1] = (int)((uint32_t)K << 16);              // tensor_dim0[15:0]
  g1d[2] = (int)(((uint32_t)K >> 16) & 0xFFFFu) | // tensor_dim0[31:16]
           (int)(0x0u << 16);                     // tensor_dim1[15:0] (of 1<<20)
  g1d[3] = (int)0x10u |                           // tensor_dim1[31:16] = 16 -> 1<<20
           (int)(32u << 16);                      // tile_dim0 = 32 elements
  g1d[4] = 128;                                   // tile_dim1 = 128, tile_dim2 = 0
  g1d[5] = ldb;                                   // tensor_dim0_stride (elements)
  g1d[6] = 0;
  g1d[7] = 0;
  const mb_i32x4 zero4 = {0, 0, 0, 0};
  const mb_i32x8 zero8 = {0, 0, 0, 0, 0, 0, 0, 0};

  auto issueA = [&](int s, int buf) {
    const uint64_t koff = (uint64_t)s * 32 * sizeof(__bf16);
    const uint32_t boff = (uint32_t)buf * (128 * 32 * sizeof(__bf16));
    MB_ASYNC_CP16(la0 + boff,        ga0 + koff);
    MB_ASYNC_CP16(la0 + boff + 4096, ga1 + koff);
  };
  auto issueB = [&](int s, int buf) {            // wave 0 only
    const uint64_t g = gbB + (uint64_t)s * 32 * sizeof(__bf16);
    mb_u32x4 g0;
    g0[0] = 1u;                                  // count=1, user descriptor
    g0[1] = ldsB + (uint32_t)buf * 8192u;        // lds_addr
    g0[2] = (uint32_t)g;                         // global_addr[31:0]
    g0[3] = ((uint32_t)(g >> 32) & 0x1FFFFFFu) | // global_addr[56:32]
            0x80000000u;                         // type = 2 ("image")
    __builtin_amdgcn_tensor_load_to_lds(g0, g1d, zero4, zero4, zero8, 0);
  };

  f32x8 acc[2][4];
#pragma unroll
  for (int rt = 0; rt < 2; ++rt)
#pragma unroll
    for (int ct = 0; ct < 4; ++ct)
#pragma unroll
      for (int r = 0; r < 8; ++r) acc[rt][ct][r] = 0.0f;

  const int S = K >> 5;                      // K % 32 == 0 at all call sites
  issueA(0, 0);
  if (wave == 0) issueB(0, 0);

  for (int s = 0; s < S; ++s) {
    const int buf = s & 1;
    MB_WAIT_ASYNC();                         // own A copies of tile s done
    __builtin_amdgcn_s_wait_tensorcnt(0);    // wave0's TDM done (others: no-op)
    __syncthreads();                         // all copies visible to the WG
    if (s + 1 < S) {                         // prefetch next tile, other buffer
      issueA(s + 1, buf ^ 1);
      if (wave == 0) issueB(s + 1, buf ^ 1);
    }

    bf16x16 af[2];
#pragma unroll
    for (int rt = 0; rt < 2; ++rt)
      af[rt] = mb_lds_frag(&At[buf][wr * 32 + rt * 16 + l16][0], kbase);
    bf16x16 bfm[4];
#pragma unroll
    for (int ct = 0; ct < 4; ++ct)
      bfm[ct] = mb_lds_frag(&Bts[buf][wc * 64 + ct * 16 + l16][0], kbase);
#pragma unroll
    for (int rt = 0; rt < 2; ++rt)
#pragma unroll
      for (int ct = 0; ct < 4; ++ct)
        acc[rt][ct] = __builtin_amdgcn_wmma_f32_16x16x32_bf16(
            false, af[rt], false, bfm[ct], (short)0, acc[rt][ct], false, false);
    // no trailing barrier: next iteration's copies target the other buffer,
    // and the barrier at its top orders buffer reuse.
  }

  // epilogue: D layout -> VGPR r = row r (+8 for lanes 16..31), col = lane%16
#pragma unroll
  for (int rt = 0; rt < 2; ++rt) {
#pragma unroll
    for (int ct = 0; ct < 4; ++ct) {
      const int growBase = blockM + wr * 32 + rt * 16 + laneHalf * 8;
      const int gcol     = blockN + wc * 64 + ct * 16 + l16;
      if (gcol < N) {
        const float bv = bias ? bias[gcol] : 0.0f;
#pragma unroll
        for (int r = 0; r < 8; ++r) {
          const int grow = growBase + r;
          float v = acc[rt][ct][r] + bv;
          if (act == 1) v = mb_softplus(v);
          C[(size_t)grow * ldc + gcol] = v;
          if (Cb) Cb[(size_t)grow * ldcb + gcol] = (__bf16)v;
        }
      }
    }
  }
}

// f32 -> bf16 elementwise
__global__ __launch_bounds__(256)
void mb_cvt_bf16(const float* __restrict__ in, __bf16* __restrict__ out, int n) {
  const int idx = blockIdx.x * blockDim.x + threadIdx.x;
  if (idx < n) out[idx] = (__bf16)in[idx];
}

// transpose f32 [K][N] -> bf16 [N][K] (weights are small; one shot per call)
__global__ __launch_bounds__(256)
void mb_transpose_bf16(const float* __restrict__ in, __bf16* __restrict__ out,
                       int K, int N) {
  const int idx = blockIdx.x * blockDim.x + threadIdx.x;
  if (idx >= N * K) return;
  const int n = idx / K;
  const int k = idx - n * K;
  out[idx] = (__bf16)in[(size_t)k * N + n];
}

// causal depthwise conv1d (K=4, left pad 3) + bias + SiLU; f32 + bf16 outputs
__global__ __launch_bounds__(256)
void mb_conv_silu(const float* __restrict__ xz, const float* __restrict__ w,
                  const float* __restrict__ b, float* __restrict__ xc,
                  __bf16* __restrict__ xcb) {
  const int idx = blockIdx.x * blockDim.x + threadIdx.x;
  if (idx >= MB_ROWS * MB_DINNER) return;
  const int d  = idx & (MB_DINNER - 1);
  const int l  = (idx / MB_DINNER) & (MB_L - 1);
  const int bb = idx / (MB_DINNER * MB_L);
  float acc = b[d];
#pragma unroll
  for (int k = 0; k < 4; ++k) {
    const int lt = l - 3 + k;
    if (lt >= 0)
      acc += w[d * 4 + k] *
             xz[((size_t)(bb * MB_L + lt)) * (2 * MB_DINNER) + d];
  }
  const float s = acc * (1.0f / (1.0f + __expf(-acc)));   // SiLU
  xc[idx]  = s;
  xcb[idx] = (__bf16)s;
}

// A = -exp(A_log)
__global__ __launch_bounds__(256)
void mb_prep_A(const float* __restrict__ A_log, float* __restrict__ A) {
  const int idx = blockIdx.x * blockDim.x + threadIdx.x;
  if (idx < MB_DINNER * MB_DSTATE) A[idx] = -__expf(A_log[idx]);
}

// ---------------------------------------------------------------------------
// Selective scan: block = (batch, 32 channels); 8 lanes/channel, 16 states
// each in VGPRs (128 blocks -> better occupancy, shorter v_exp chains).
// One barrier per step via double-buffered B_t/C_t slot.
// ---------------------------------------------------------------------------
__global__ __launch_bounds__(256)
void mb_scan(const float* __restrict__ dt, const float* __restrict__ xc,
             const float* __restrict__ xdbl, const float* __restrict__ A,
             float* __restrict__ y) {
  __shared__ float BC[2][2 * MB_DSTATE];
  const int b    = blockIdx.x >> 6;
  const int dblk = blockIdx.x & 63;
  const int tid  = threadIdx.x;
  const int c    = tid >> 3;                 // 0..31 channel in block
  const int q    = tid & 7;                  // state octant
  const int d    = dblk * 32 + c;

  float h[16], a[16];
#pragma unroll
  for (int i = 0; i < 16; ++i) {
    h[i] = 0.0f;
    a[i] = A[(size_t)d * MB_DSTATE + q * 16 + i];
  }

  for (int t = 0; t < MB_L; ++t) {
    const size_t row = (size_t)b * MB_L + t;
    const int slot = t & 1;
    BC[slot][tid] = xdbl[row * MB_XPROJ_N + MB_DTRANK + tid];
    __syncthreads();

    const float sdt = dt[row * MB_DINNER + d];
    const float su  = xc[row * MB_DINNER + d];
    const float dtu = sdt * su;
    float part = 0.0f;
#pragma unroll
    for (int i = 0; i < 16; ++i) {
      const int n = q * 16 + i;
      const float dA = __expf(sdt * a[i]);
      h[i] = dA * h[i] + dtu * BC[slot][n];
      part += h[i] * BC[slot][MB_DSTATE + n];
    }
    part += __shfl_xor(part, 1, 32);
    part += __shfl_xor(part, 2, 32);
    part += __shfl_xor(part, 4, 32);
    if (q == 0) y[row * MB_DINNER + d] = part;
    // no second barrier: next step writes the other slot
  }
}

// y_bf16 = (y_scan + x_conv * D) * SiLU(z)
__global__ __launch_bounds__(256)
void mb_gate(const float* __restrict__ xz, const float* __restrict__ xc,
             const float* __restrict__ Dp, const float* __restrict__ y,
             __bf16* __restrict__ yb) {
  const int idx = blockIdx.x * blockDim.x + threadIdx.x;
  if (idx >= MB_ROWS * MB_DINNER) return;
  const int d = idx & (MB_DINNER - 1);
  const size_t row = (size_t)idx / MB_DINNER;
  const float z  = xz[row * (2 * MB_DINNER) + MB_DINNER + d];
  const float sz = z * (1.0f / (1.0f + __expf(-z)));
  yb[idx] = (__bf16)((y[idx] + xc[idx] * Dp[d]) * sz);
}

// ---------------------------------------------------------------------------
extern "C" void kernel_launch(void* const* d_in, const int* in_sizes, int n_in,
                              void* d_out, int out_size, void* d_ws, size_t ws_size,
                              hipStream_t stream) {
  const float* hid    = (const float*)d_in[0];
  const float* W_in   = (const float*)d_in[1];
  const float* conv_w = (const float*)d_in[2];
  const float* conv_b = (const float*)d_in[3];
  const float* W_xprj = (const float*)d_in[4];
  const float* W_dt   = (const float*)d_in[5];
  const float* b_dt   = (const float*)d_in[6];
  const float* A_log  = (const float*)d_in[7];
  const float* Dp     = (const float*)d_in[8];
  const float* W_out  = (const float*)d_in[9];
  float* out = (float*)d_out;

  // ---- workspace layout ----
  float* wsf   = (float*)d_ws;
  float* xz    = wsf;                                      // 4096*4096
  float* xconv = xz    + (size_t)MB_ROWS * 2 * MB_DINNER;  // 4096*2048
  float* xdbl  = xconv + (size_t)MB_ROWS * MB_DINNER;      // 4096*320
  float* dtb   = xdbl  + (size_t)MB_ROWS * MB_XPROJ_N;     // 4096*2048
  float* yb    = dtb   + (size_t)MB_ROWS * MB_DINNER;      // 4096*2048
  float* Aa    = yb    + (size_t)MB_ROWS * MB_DINNER;      // 2048*128
  __bf16* wsb  = (__bf16*)(Aa + (size_t)MB_DINNER * MB_DSTATE);
  __bf16* hidb   = wsb;                                          // 4096*1024
  __bf16* xconvb = hidb   + (size_t)MB_ROWS * MB_DMODEL;         // 4096*2048
  __bf16* xdblb  = xconvb + (size_t)MB_ROWS * MB_DINNER;         // 4096*320
  __bf16* ybf    = xdblb  + (size_t)MB_ROWS * MB_XPROJ_N;        // 4096*2048
  __bf16* WinT   = ybf    + (size_t)MB_ROWS * MB_DINNER;         // 4096*1024
  __bf16* WxT    = WinT   + (size_t)(2 * MB_DINNER) * MB_DMODEL; // 384*2048 (pad)
  __bf16* WdtT   = WxT    + (size_t)MB_XPROJ_NPAD * MB_DINNER;   // 2048*64
  __bf16* WoutT  = WdtT   + (size_t)MB_DINNER * MB_DTRANK;       // 1024*2048

  const dim3 blk(256);
  auto g1 = [](int n) { return dim3((n + 255) / 256); };
  auto g2 = [](int M, int N) { return dim3((N + 127) / 128, (M + 127) / 128); };

  // ---- bf16 conversions / weight transposes (tiny, memory-bound) ----
  mb_cvt_bf16<<<g1(MB_ROWS * MB_DMODEL), blk, 0, stream>>>(hid, hidb, MB_ROWS * MB_DMODEL);
  mb_transpose_bf16<<<g1(2 * MB_DINNER * MB_DMODEL), blk, 0, stream>>>(
      W_in, WinT, MB_DMODEL, 2 * MB_DINNER);
  mb_transpose_bf16<<<g1(MB_XPROJ_N * MB_DINNER), blk, 0, stream>>>(
      W_xprj, WxT, MB_DINNER, MB_XPROJ_N);
  mb_transpose_bf16<<<g1(MB_DINNER * MB_DTRANK), blk, 0, stream>>>(
      W_dt, WdtT, MB_DTRANK, MB_DINNER);
  mb_transpose_bf16<<<g1(MB_DMODEL * MB_DINNER), blk, 0, stream>>>(
      W_out, WoutT, MB_DINNER, MB_DMODEL);
  mb_prep_A<<<g1(MB_DINNER * MB_DSTATE), blk, 0, stream>>>(A_log, Aa);

  // 1) xz = hidden @ W_in            (4096 x 1024 x 4096)
  mb_gemm_bf16<<<g2(MB_ROWS, 2 * MB_DINNER), blk, 0, stream>>>(
      hidb, WinT, xz, nullptr, MB_ROWS, 2 * MB_DINNER, MB_DMODEL,
      MB_DMODEL, MB_DMODEL, 2 * MB_DINNER, 0, nullptr, 0);

  // 2) depthwise causal conv + SiLU (f32 + bf16)
  mb_conv_silu<<<g1(MB_ROWS * MB_DINNER), blk, 0, stream>>>(
      xz, conv_w, conv_b, xconv, xconvb);

  // 3) x_dbl = x_conv @ W_xproj      (4096 x 2048 x 320, ragged N)
  mb_gemm_bf16<<<g2(MB_ROWS, MB_XPROJ_N), blk, 0, stream>>>(
      xconvb, WxT, xdbl, xdblb, MB_ROWS, MB_XPROJ_N, MB_DINNER,
      MB_DINNER, MB_DINNER, MB_XPROJ_N, MB_XPROJ_N, nullptr, 0);

  // 4) dt = softplus(x_dbl[:, :64] @ W_dt + b_dt)   (4096 x 64 x 2048)
  mb_gemm_bf16<<<g2(MB_ROWS, MB_DINNER), blk, 0, stream>>>(
      xdblb, WdtT, dtb, nullptr, MB_ROWS, MB_DINNER, MB_DTRANK,
      MB_XPROJ_N, MB_DTRANK, MB_DINNER, 0, b_dt, 1);

  // 5) selective scan (128 blocks = 2 batches x 64 channel groups)
  mb_scan<<<MB_BSZ * (MB_DINNER / 32), blk, 0, stream>>>(dtb, xconv, xdbl, Aa, yb);

  // 6) gate -> bf16
  mb_gate<<<g1(MB_ROWS * MB_DINNER), blk, 0, stream>>>(xz, xconv, Dp, yb, ybf);

  // 7) out = y @ W_out               (4096 x 2048 x 1024)
  mb_gemm_bf16<<<g2(MB_ROWS, MB_DMODEL), blk, 0, stream>>>(
      ybf, WoutT, out, nullptr, MB_ROWS, MB_DMODEL, MB_DINNER,
      MB_DINNER, MB_DINNER, MB_DMODEL, 0, nullptr, 0);
}
